// CustomMultiMarginLoss_36610301231580
// MI455X (gfx1250) — compile-verified, matrix operand order checked by reading, not run on hardware
//
#include <hip/hip_runtime.h>
#include <hip/hip_fp16.h>

// MultiMarginLoss (p=1, margin=1), mean reduction.
// B=8192 rows, C=32000 classes, fp32 input, int64 target -> scalar f32.
//
// Memory-bound: 1.05 GB single-pass stream, ~45us floor at 23.3 TB/s.
// One wave per row; b128 non-temporal loads; hinge = max(x + (margin-correct), 0);
// accumulation offloaded to the matrix pipe via WMMA with an all-ones B matrix
// (grand sum of all A entries lands in C, exact in f32). Target column handled
// analytically by subtracting MARGIN once per row.

typedef float    v2f  __attribute__((ext_vector_type(2)));
typedef float    v4f  __attribute__((ext_vector_type(4)));
typedef float    v8f  __attribute__((ext_vector_type(8)));
typedef _Float16 v16h __attribute__((ext_vector_type(16)));

#define MARGIN_F 1.0f
#define WAVES_PER_BLOCK 8

__global__ __launch_bounds__(256) void mml_rows_kernel(
    const float* __restrict__ in, const long long* __restrict__ tgt,
    float* __restrict__ row_loss, int Brows, int C)
{
#if defined(__HIP_DEVICE_COMPILE__)
    const int wave = threadIdx.x >> 5;
    const int lane = threadIdx.x & 31;
    const int row  = blockIdx.x * WAVES_PER_BLOCK + wave;
    if (row >= Brows) return;

    const float* __restrict__ rp = in + (size_t)row * (size_t)C;
    const float correct = rp[tgt[row]];       // same address per wave -> 1 request
    const float s = MARGIN_F - correct;       // hinge = max(x + s, 0)

    const int nIter = C >> 7;                 // 128 columns per wave iteration
    const v4f* __restrict__ p4 = (const v4f*)rp;
    float wsum;

#if __has_builtin(__builtin_amdgcn_wmma_f32_16x16x4_f32)
    // ---- exact f32 accumulation on the matrix pipe ----
    v8f acc0 = {};
    v8f acc1 = {};
    v2f bones; bones.x = 1.0f; bones.y = 1.0f;           // B = all ones (4x16)
    int i4 = lane;
    for (int it = 0; it < nIter; ++it, i4 += 32) {
        v4f x = __builtin_nontemporal_load(p4 + i4);     // global_load_b128, TH=NT
        v2f a01, a23;
        a01.x = fmaxf(x.x + s, 0.0f);
        a01.y = fmaxf(x.y + s, 0.0f);
        a23.x = fmaxf(x.z + s, 0.0f);
        a23.y = fmaxf(x.w + s, 0.0f);
        // C[M][N] += sum_K A[M][K]  (B all ones): accumulates every A entry.
        acc0 = __builtin_amdgcn_wmma_f32_16x16x4_f32(false, a01, false, bones,
                                                     (short)0, acc0, false, false);
        acc1 = __builtin_amdgcn_wmma_f32_16x16x4_f32(false, a23, false, bones,
                                                     (short)0, acc1, false, false);
    }
    v8f acc = acc0 + acc1;
    // Lane L (0..15) holds rows 0..7 sums (replicated across columns) in acc[0..7];
    // lanes 16..31 hold rows 8..15. Per-lane column sum, then take lane 0 + lane 16.
    float partial = ((acc[0] + acc[1]) + (acc[2] + acc[3]))
                  + ((acc[4] + acc[5]) + (acc[6] + acc[7]));
    wsum = __shfl(partial, 0, 32) + __shfl(partial, 16, 32);
#else
    // ---- fallback: codegen-confirmed f16 WMMA (f32 accumulate) ----
    v8f acc = {};
    v16h bones;
    #pragma unroll
    for (int i = 0; i < 16; ++i) bones[i] = (_Float16)1.0f;
    int i4 = lane;
    const int nMac = nIter >> 2;              // 4 b128 loads feed one 16-wide A
    for (int mt = 0; mt < nMac; ++mt) {
        v16h a;
        #pragma unroll
        for (int j = 0; j < 4; ++j) {
            v4f x = __builtin_nontemporal_load(p4 + i4); i4 += 32;
            a[4*j+0] = (_Float16)fmaxf(x.x + s, 0.0f);
            a[4*j+1] = (_Float16)fmaxf(x.y + s, 0.0f);
            a[4*j+2] = (_Float16)fmaxf(x.z + s, 0.0f);
            a[4*j+3] = (_Float16)fmaxf(x.w + s, 0.0f);
        }
        acc = __builtin_amdgcn_wmma_f32_16x16x32_f16(false, a, false, bones,
                                                     (short)0, acc, false, false);
    }
    const int rem = nIter & 3;
    if (rem) {
        v16h a;
        #pragma unroll
        for (int i = 0; i < 16; ++i) a[i] = (_Float16)0.0f;
        {   v4f x = __builtin_nontemporal_load(p4 + i4); i4 += 32;
            a[0] = (_Float16)fmaxf(x.x + s, 0.0f); a[1] = (_Float16)fmaxf(x.y + s, 0.0f);
            a[2] = (_Float16)fmaxf(x.z + s, 0.0f); a[3] = (_Float16)fmaxf(x.w + s, 0.0f); }
        if (rem > 1) {
            v4f x = __builtin_nontemporal_load(p4 + i4); i4 += 32;
            a[4] = (_Float16)fmaxf(x.x + s, 0.0f); a[5] = (_Float16)fmaxf(x.y + s, 0.0f);
            a[6] = (_Float16)fmaxf(x.z + s, 0.0f); a[7] = (_Float16)fmaxf(x.w + s, 0.0f); }
        if (rem > 2) {
            v4f x = __builtin_nontemporal_load(p4 + i4); i4 += 32;
            a[8]  = (_Float16)fmaxf(x.x + s, 0.0f); a[9]  = (_Float16)fmaxf(x.y + s, 0.0f);
            a[10] = (_Float16)fmaxf(x.z + s, 0.0f); a[11] = (_Float16)fmaxf(x.w + s, 0.0f); }
        acc = __builtin_amdgcn_wmma_f32_16x16x32_f16(false, a, false, bones,
                                                     (short)0, acc, false, false);
    }
    float partial = ((acc[0] + acc[1]) + (acc[2] + acc[3]))
                  + ((acc[4] + acc[5]) + (acc[6] + acc[7]));
    wsum = __shfl(partial, 0, 32) + __shfl(partial, 16, 32);
#endif

    // Scalar tail for C not divisible by 128 (empty for C=32000).
    float extra = 0.0f;
    for (int c = (nIter << 7) + lane; c < C; c += 32)
        extra += fmaxf(rp[c] + s, 0.0f);
    #pragma unroll
    for (int off = 16; off > 0; off >>= 1)
        extra += __shfl_xor(extra, off, 32);

    // Target column contributed exactly max(margin,0)^1 = MARGIN; remove it.
    if (lane == 0) row_loss[row] = (wsum + extra) - MARGIN_F;
#endif // __HIP_DEVICE_COMPILE__
}

__global__ __launch_bounds__(256) void mml_reduce_kernel(
    const float* __restrict__ row_loss, float* __restrict__ out, int n)
{
    __shared__ float sm[256];
    float s = 0.0f;
    for (int i = threadIdx.x; i < n; i += 256) s += row_loss[i];
    sm[threadIdx.x] = s;
    __syncthreads();
    for (int off = 128; off > 0; off >>= 1) {
        if ((int)threadIdx.x < off) sm[threadIdx.x] += sm[threadIdx.x + off];
        __syncthreads();
    }
    if (threadIdx.x == 0) out[0] = sm[0] / (float)n;
}

extern "C" void kernel_launch(void* const* d_in, const int* in_sizes, int n_in,
                              void* d_out, int out_size, void* d_ws, size_t ws_size,
                              hipStream_t stream)
{
    const float*     in  = (const float*)d_in[0];
    const long long* tgt = (const long long*)d_in[1];   // jnp.int64 target
    const int Brows = in_sizes[1];
    const int C     = in_sizes[0] / Brows;

    float* row_loss = (float*)d_ws;                     // Brows * 4 bytes scratch

    const int blocks = (Brows + WAVES_PER_BLOCK - 1) / WAVES_PER_BLOCK;
    mml_rows_kernel<<<blocks, 32 * WAVES_PER_BLOCK, 0, stream>>>(in, tgt, row_loss, Brows, C);
    mml_reduce_kernel<<<1, 256, 0, stream>>>(row_loss, (float*)d_out, Brows);
}
